// TreeRegressor_20572893348711
// MI455X (gfx1250) — compile-verified, matrix operand order checked by reading, not run on hardware
//
#include <hip/hip_runtime.h>
#include <hip/hip_bf16.h>

// Types for WMMA
typedef _Float16 half4_t __attribute__((ext_vector_type(4)));
typedef _Float16 half8_t __attribute__((ext_vector_type(8)));
typedef _Float16 v16h    __attribute__((ext_vector_type(16)));
typedef float    v8f     __attribute__((ext_vector_type(8)));

// ---------------------------------------------------------------------------
// Fragment loader for 16-bit A/B operands of V_WMMA_F32_16X16X32_F16.
// LDS tile is row-major [rows][128] f16 (for B we pre-transpose weights so
// "row" = output column and K runs contiguously).
// Per ISA 7.12.2 (16-bit A 16x32):
//   lane L (0..15):  row M=L,  halves 0..7 = K k0..k0+7,  8..15 = K k0+16..k0+23
//   lane L (16..31): row M=L-16, same with K offset +8.
// ---------------------------------------------------------------------------
__device__ __forceinline__ v16h frag128(const _Float16* base, int rc, int k0, int hi) {
    const _Float16* p = base + rc * 128 + k0 + 8 * hi;
    half8_t lo = *(const half8_t*)(p);        // ds_load_b128
    half8_t hv = *(const half8_t*)(p + 16);
    v16h r;
#pragma unroll
    for (int i = 0; i < 8; ++i) { r[i] = lo[i]; r[i + 8] = hv[i]; }
    return r;
}

// ---------------------------------------------------------------------------
// Convert 5 [128][128] f32 weights to f16, transposed to [out][in].
// ---------------------------------------------------------------------------
__global__ void prep_weights(const float* w0, const float* w1, const float* w2,
                             const float* w3, const float* w4,
                             _Float16* o0, _Float16* o1, _Float16* o2,
                             _Float16* o3, _Float16* o4) {
    int i = blockIdx.x * 256 + threadIdx.x;   // 0..16383
    int n = i & 127, k = i >> 7;
    const float* src; _Float16* dst;
    switch (blockIdx.y) {
        case 0: src = w0; dst = o0; break;
        case 1: src = w1; dst = o1; break;
        case 2: src = w2; dst = o2; break;
        case 3: src = w3; dst = o3; break;
        default: src = w4; dst = o4; break;
    }
    dst[n * 128 + k] = (_Float16)src[k * 128 + n];
}

__global__ void copy_f32v4(const float* __restrict__ in, float* __restrict__ out, int n4) {
    int i = blockIdx.x * 256 + threadIdx.x;
    if (i < n4) ((float4*)out)[i] = ((const float4*)in)[i];
}

__global__ void zero_f32(float* p, int n) {
    int i = blockIdx.x * 256 + threadIdx.x;
    if (i < n) p[i] = 0.0f;
}

// agg[dst] += h[src] over edges; one wave (32 lanes x float4) per edge.
__global__ void scatter_edges(const float* __restrict__ h, const int* __restrict__ ei,
                              float* __restrict__ agg, int ne) {
    int t = blockIdx.x * 256 + threadIdx.x;
    int e = t >> 5, lane = t & 31;
    if (e >= ne) return;
    int s = ei[e];
    int d = ei[ne + e];
    float4 v = *(const float4*)(h + (size_t)s * 128 + lane * 4);
    float* dp = agg + (size_t)d * 128 + lane * 4;
    unsafeAtomicAdd(dp + 0, v.x);
    unsafeAtomicAdd(dp + 1, v.y);
    unsafeAtomicAdd(dp + 2, v.z);
    unsafeAtomicAdd(dp + 3, v.w);
}

// Per-graph sums + counts for mean pooling.
__global__ void pool_scatter(const float* __restrict__ h, const int* __restrict__ xb,
                             float* __restrict__ sums, float* __restrict__ cnts, int n) {
    int t = blockIdx.x * 256 + threadIdx.x;
    int i = t >> 5, lane = t & 31;
    if (i >= n) return;
    int g = xb[i];
    float4 v = *(const float4*)(h + (size_t)i * 128 + lane * 4);
    float* dp = sums + (size_t)g * 128 + lane * 4;
    unsafeAtomicAdd(dp + 0, v.x);
    unsafeAtomicAdd(dp + 1, v.y);
    unsafeAtomicAdd(dp + 2, v.z);
    unsafeAtomicAdd(dp + 3, v.w);
    if (lane == 0) unsafeAtomicAdd(cnts + g, 1.0f);
}

// ---------------------------------------------------------------------------
// Fused MLP: out = relu(agg @ W1 + b1) @ W2 + b2, 64 rows per block.
// 8 waves; each wave owns a 16-column slice and 4 M-tiles (16 rows each),
// reusing each staged B fragment across 4 WMMAs. Both 128x128 f16 weights
// resident in LDS (~98KB total of the 320KB/WGP).
// ---------------------------------------------------------------------------
__global__ __launch_bounds__(256) void fused_mlp(const float* __restrict__ aggIn,
                                                 const _Float16* __restrict__ W1t,
                                                 const float* __restrict__ b1,
                                                 const _Float16* __restrict__ W2t,
                                                 const float* __restrict__ b2,
                                                 float* __restrict__ out, int nrows) {
    __shared__ _Float16 sW1[128 * 128];
    __shared__ _Float16 sW2[128 * 128];
    __shared__ _Float16 sA[64 * 128];
    __shared__ _Float16 sT[64 * 128];
    __shared__ float sB1[128];
    __shared__ float sB2[128];

    const int tid = threadIdx.x;
    const int row0 = blockIdx.x * 64;

    // Stage weights (32KB each) into LDS as uint4 copies.
    {
        const uint4* s1 = (const uint4*)W1t; uint4* d1 = (uint4*)sW1;
        const uint4* s2 = (const uint4*)W2t; uint4* d2 = (uint4*)sW2;
#pragma unroll
        for (int i = 0; i < 8; ++i) {
            d1[tid + 256 * i] = s1[tid + 256 * i];
            d2[tid + 256 * i] = s2[tid + 256 * i];
        }
    }
    if (tid < 128) { sB1[tid] = b1[tid]; sB2[tid] = b2[tid]; }

    // Stage A tile: 64x128 f32 -> f16 (2048 float4 loads, half4 LDS stores).
    {
        const float4* a4 = (const float4*)(aggIn + (size_t)row0 * 128);
        half4_t* s4 = (half4_t*)sA;
#pragma unroll
        for (int i = 0; i < 8; ++i) {
            int idx = tid + 256 * i;             // 0..2047 (32 float4 per row)
            int r = row0 + (idx >> 5);
            float4 v = make_float4(0.f, 0.f, 0.f, 0.f);
            if (r < nrows) v = a4[idx];
            half4_t h;
            h[0] = (_Float16)v.x; h[1] = (_Float16)v.y;
            h[2] = (_Float16)v.z; h[3] = (_Float16)v.w;
            s4[idx] = h;                          // ds_store_b64
        }
    }
    __syncthreads();

    const int lane = tid & 31, wave = tid >> 5;
    const int nloc = lane & 15, hi = lane >> 4;
    const int col = wave * 16 + nloc;

    // GEMM1: t = relu(A @ W1 + b1)
    v8f acc[4] = {{}, {}, {}, {}};
#pragma unroll
    for (int k0 = 0; k0 < 128; k0 += 32) {
        v16h b = frag128(sW1, col, k0, hi);
#pragma unroll
        for (int mt = 0; mt < 4; ++mt) {
            v16h a = frag128(sA + mt * 16 * 128, nloc, k0, hi);
            acc[mt] = __builtin_amdgcn_wmma_f32_16x16x32_f16(false, a, false, b,
                                                             (short)0, acc[mt], false, false);
        }
    }
    float bb1 = sB1[col];
#pragma unroll
    for (int mt = 0; mt < 4; ++mt) {
#pragma unroll
        for (int r = 0; r < 8; ++r) {
            int m = mt * 16 + r + 8 * hi;        // C layout: vgpr r, lanes>=16 -> M+8
            float v = acc[mt][r] + bb1;
            sT[m * 128 + col] = (_Float16)(v > 0.0f ? v : 0.0f);
        }
    }
    __syncthreads();

    // GEMM2: out = t @ W2 + b2
    v8f acc2[4] = {{}, {}, {}, {}};
#pragma unroll
    for (int k0 = 0; k0 < 128; k0 += 32) {
        v16h b = frag128(sW2, col, k0, hi);
#pragma unroll
        for (int mt = 0; mt < 4; ++mt) {
            v16h a = frag128(sT + mt * 16 * 128, nloc, k0, hi);
            acc2[mt] = __builtin_amdgcn_wmma_f32_16x16x32_f16(false, a, false, b,
                                                              (short)0, acc2[mt], false, false);
        }
    }
    float bb2 = sB2[col];
#pragma unroll
    for (int mt = 0; mt < 4; ++mt) {
#pragma unroll
        for (int r = 0; r < 8; ++r) {
            int m = mt * 16 + r + 8 * hi;
            if (row0 + m < nrows)
                out[(size_t)(row0 + m) * 128 + col] = acc2[mt][r] + bb2;
        }
    }
}

// ---------------------------------------------------------------------------
// Regressor head: pooled = sums/max(cnt,1); t = relu(pooled@W1+b1);
// out = t @ W2(128x1) + b2. 16 graphs per block (only 32 blocks total).
// ---------------------------------------------------------------------------
__global__ __launch_bounds__(256) void reg_head(const float* __restrict__ sums,
                                                const float* __restrict__ cnts,
                                                const _Float16* __restrict__ W1t,
                                                const float* __restrict__ b1,
                                                const float* __restrict__ W2,
                                                const float* __restrict__ b2,
                                                float* __restrict__ out, int ngraphs) {
    __shared__ _Float16 sW1[128 * 128];
    __shared__ _Float16 sA[16 * 128];
    __shared__ float sTf[16 * 128];
    __shared__ float sB1[128];

    const int tid = threadIdx.x;
    const int row0 = blockIdx.x * 16;

    {
        const uint4* s1 = (const uint4*)W1t; uint4* d1 = (uint4*)sW1;
#pragma unroll
        for (int i = 0; i < 8; ++i) d1[tid + 256 * i] = s1[tid + 256 * i];
    }
    if (tid < 128) sB1[tid] = b1[tid];
#pragma unroll
    for (int i = 0; i < 8; ++i) {
        int idx = tid + 256 * i;
        int g = row0 + (idx >> 7);
        float v = 0.0f;
        if (g < ngraphs) {
            float cn = cnts[g];
            cn = cn < 1.0f ? 1.0f : cn;
            v = sums[(size_t)row0 * 128 + idx] / cn;
        }
        sA[idx] = (_Float16)v;
    }
    __syncthreads();

    const int lane = tid & 31, wave = tid >> 5;
    const int nloc = lane & 15, hi = lane >> 4;
    const int col = wave * 16 + nloc;

    v8f c = {};
#pragma unroll
    for (int k0 = 0; k0 < 128; k0 += 32) {
        v16h a = frag128(sA, nloc, k0, hi);
        v16h b = frag128(sW1, col, k0, hi);
        c = __builtin_amdgcn_wmma_f32_16x16x32_f16(false, a, false, b, (short)0, c, false, false);
    }
    float bb1 = sB1[col];
#pragma unroll
    for (int r = 0; r < 8; ++r) {
        int m = r + 8 * hi;
        float v = c[r] + bb1;
        sTf[m * 128 + col] = v > 0.0f ? v : 0.0f;
    }
    __syncthreads();

    if (tid < 16 && (row0 + tid) < ngraphs) {
        float s = b2[0];
        const float* tr = sTf + tid * 128;
        for (int k = 0; k < 128; ++k) s += tr[k] * W2[k];
        out[row0 + tid] = s;
    }
}

// ---------------------------------------------------------------------------
extern "C" void kernel_launch(void* const* d_in, const int* in_sizes, int n_in,
                              void* d_out, int out_size, void* d_ws, size_t ws_size,
                              hipStream_t stream) {
    const float* x     = (const float*)d_in[0];
    const int*   ei    = (const int*)d_in[1];
    /* d_in[2] = pos, unused */
    const int*   xb    = (const int*)d_in[3];
    const float* c0W1  = (const float*)d_in[4];
    const float* c0b1  = (const float*)d_in[5];
    const float* c0W2  = (const float*)d_in[6];
    const float* c0b2  = (const float*)d_in[7];
    const float* c1W1  = (const float*)d_in[8];
    const float* c1b1  = (const float*)d_in[9];
    const float* c1W2  = (const float*)d_in[10];
    const float* c1b2  = (const float*)d_in[11];
    const float* rW1   = (const float*)d_in[12];
    const float* rb1   = (const float*)d_in[13];
    const float* rW2   = (const float*)d_in[14];
    const float* rb2   = (const float*)d_in[15];
    float* outp = (float*)d_out;

    const int N = in_sizes[0] / 128;
    const int E = in_sizes[1] / 2;
    const int G = out_size;

    // Workspace carve-up (f32 region first keeps 16B alignment for f16 weights).
    float* agg  = (float*)d_ws;                    // N*128
    float* hbuf = agg + (size_t)N * 128;           // N*128
    float* sums = hbuf + (size_t)N * 128;          // G*128
    float* cnts = sums + (size_t)G * 128;          // G
    _Float16* w16 = (_Float16*)(cnts + G);
    _Float16* c0W1t = w16;
    _Float16* c0W2t = w16 + 16384;
    _Float16* c1W1t = w16 + 2 * 16384;
    _Float16* c1W2t = w16 + 3 * 16384;
    _Float16* rW1t  = w16 + 4 * 16384;

    const int n4 = N * 32;                          // float4 count of N*128
    const int mlpBlocks = (N + 63) / 64;
    const int regBlocks = (G + 15) / 16;

    prep_weights<<<dim3(64, 5), 256, 0, stream>>>(c0W1, c0W2, c1W1, c1W2, rW1,
                                                  c0W1t, c0W2t, c1W1t, c1W2t, rW1t);

    // conv0: agg = x + scatter_add(x[src] -> dst); h = MLP(agg)
    copy_f32v4<<<(n4 + 255) / 256, 256, 0, stream>>>(x, agg, n4);
    scatter_edges<<<(E * 32 + 255) / 256, 256, 0, stream>>>(x, ei, agg, E);
    fused_mlp<<<mlpBlocks, 256, 0, stream>>>(agg, c0W1t, c0b1, c0W2t, c0b2, hbuf, N);

    // conv1
    copy_f32v4<<<(n4 + 255) / 256, 256, 0, stream>>>(hbuf, agg, n4);
    scatter_edges<<<(E * 32 + 255) / 256, 256, 0, stream>>>(hbuf, ei, agg, E);
    fused_mlp<<<mlpBlocks, 256, 0, stream>>>(agg, c1W1t, c1b1, c1W2t, c1b2, hbuf, N);

    // mean pool + regressor head
    zero_f32<<<(G * 128 + G + 255) / 256, 256, 0, stream>>>(sums, G * 128 + G);
    pool_scatter<<<(N * 32 + 255) / 256, 256, 0, stream>>>(hbuf, xb, sums, cnts, N);
    reg_head<<<regBlocks, 256, 0, stream>>>(sums, cnts, rW1t, rb1, rW2, rb2, outp, G);
}
